// SimpleFPCNN3_38998303048152
// MI455X (gfx1250) — matrix operand order repaired and checked by reading, not compiled, not run on hardware
//
#include <hip/hip_runtime.h>
#include <hip/hip_bf16.h>

// MI455X (gfx1250): convs as implicit-GEMM on v_wmma_f32_16x16x32_f16.
// K ordering = tap*16 + channel so per-chunk taps are compile-time constants
// and every A/B fragment is built from contiguous ds_load_b128s.

#define HH 720
#define WW 1280
#define HW (HH * WW)

typedef __attribute__((ext_vector_type(16))) _Float16 v16h;
typedef __attribute__((ext_vector_type(8)))  _Float16 v8h;
typedef __attribute__((ext_vector_type(8)))  float    v8f;

__device__ __forceinline__ int iclamp(int v, int lo, int hi) {
    return v < lo ? lo : (v > hi ? hi : v);
}

// ---------------------------------------------------------------------------
// 3x3 SAME (zero-pad) conv, NCHW fp32 in/out, WMMA f16 16x16x32.
// Block = 128 threads (4 waves), tile = 16x16 pixels.
// GEMM: M = 16 pixels of one tile row, N = 16 out channels (COUT<=16 masked),
// K = tap*16 + cin, K_total = 9*16 = 144 -> 5 chunks of 32 (last half-empty).
// A fragment (lane half hi): elems 0..7  = tap 2ch,   channels 8hi..8hi+7
//                            elems 8..15 = tap 2ch+1, channels 8hi..8hi+7
// Both are 16B contiguous in the channel-last LDS tile -> ds_load_b128 each.
// ---------------------------------------------------------------------------
template <int CIN, int COUT, bool RELU>
__global__ __launch_bounds__(128) void conv3x3_wmma(
    const float* __restrict__ in, const float* __restrict__ wgt,
    const float* __restrict__ bias, float* __restrict__ out) {
    __shared__ __align__(32) _Float16 s_in[18][18][16];  // channel-last halo tile
    __shared__ __align__(32) _Float16 s_wT[16][160];     // B^T: [n][k], zero-padded

    const int tx0 = blockIdx.x * 16;
    const int ty0 = blockIdx.y * 16;
    const int tid = threadIdx.x;

    __builtin_prefetch(wgt, 0, 0);  // global_prefetch_b8

    // ---- stage input halo tile (f32 -> f16), channel-last, pad ch >= CIN ----
#pragma unroll
    for (int c = 0; c < 16; ++c) {
        for (int idx = tid; idx < 18 * 18; idx += 128) {
            int yy = idx / 18, xx = idx % 18;
            _Float16 v = (_Float16)0;
            if (c < CIN) {
                int gy = ty0 + yy - 1, gx = tx0 + xx - 1;
                if (gy >= 0 && gy < HH && gx >= 0 && gx < WW)
                    v = (_Float16)in[(size_t)c * HW + gy * WW + gx];
            }
            s_in[yy][xx][c] = v;
        }
    }
    // ---- stage weights transposed: s_wT[n][k], k = t*16 + c ----
    for (int idx = tid; idx < 16 * 160; idx += 128) {
        int n = idx / 160;
        int k = idx % 160;
        int t = k >> 4, c = k & 15;
        float v = 0.0f;
        if (t < 9 && c < CIN && n < COUT) v = wgt[n * (CIN * 9) + c * 9 + t];
        s_wT[n][k] = (_Float16)v;
    }
    __syncthreads();

    const int wave  = tid >> 5;
    const int lane  = tid & 31;
    const int laneM = lane & 15;
    const int hi    = lane >> 4;

    // ---- B fragments: lane half hi holds K=kb+8hi..+7 then K=kb+16+8hi..+7 ----
    v16h bfrag[5];
#pragma unroll
    for (int ch = 0; ch < 5; ++ch) {
        v8h b0 = *(const v8h*)&s_wT[laneM][ch * 32 + 8 * hi];
        v8h b1 = *(const v8h*)&s_wT[laneM][ch * 32 + 16 + 8 * hi];
        bfrag[ch] = __builtin_shufflevector(b0, b1, 0, 1, 2, 3, 4, 5, 6, 7, 8, 9,
                                            10, 11, 12, 13, 14, 15);
    }

    const float bn = (laneM < COUT) ? bias[laneM] : 0.0f;

#pragma unroll
    for (int rt = 0; rt < 4; ++rt) {
        const int row = wave + rt * 4;  // pixel row within tile (0..15)
        v8f acc = {};
#pragma unroll
        for (int ch = 0; ch < 5; ++ch) {
            const int t0 = 2 * ch;      // compile-time after unroll
            const int t1 = 2 * ch + 1;
            v8h lo = *(const v8h*)&s_in[row + t0 / 3][laneM + t0 % 3][8 * hi];
            v8h up = {};
            if (t1 < 9)
                up = *(const v8h*)&s_in[row + t1 / 3][laneM + t1 % 3][8 * hi];
            v16h a = __builtin_shufflevector(lo, up, 0, 1, 2, 3, 4, 5, 6, 7, 8, 9,
                                             10, 11, 12, 13, 14, 15);
            acc = __builtin_amdgcn_wmma_f32_16x16x32_f16(
                false, a, false, bfrag[ch], (short)0, acc, false, false);
        }
        // D: lanes 0-15: N=lane, VGPR v -> M=v; lanes 16-31: M=v+8
        if (laneM < COUT) {
            const int gy = ty0 + row;
#pragma unroll
            for (int v = 0; v < 8; ++v) {
                int col = v + 8 * hi;
                float val = acc[v] + bn;
                if (RELU) val = fmaxf(val, 0.0f);
                out[(size_t)laneM * HW + gy * WW + (tx0 + col)] = val;
            }
        }
    }
}

// ---------------------------------------------------------------------------
// LCN: per-channel 3x3 box mean/mean2 (zero SAME pad) -> rec_in:
// ch 0..2 = (x-mean)*rsqrt(var+1e-5), ch 3..5 = x
// ---------------------------------------------------------------------------
__global__ __launch_bounds__(256) void lcn_kernel(const float* __restrict__ frame,
                                                  float* __restrict__ rec) {
    __shared__ float s[3][18][20];
    const int tx0 = blockIdx.x * 16, ty0 = blockIdx.y * 16;
    const int tid = threadIdx.y * 16 + threadIdx.x;
    for (int idx = tid; idx < 3 * 18 * 18; idx += 256) {
        int c = idx / 324;
        int r = idx % 324;
        int yy = r / 18, xx = r % 18;
        int gy = ty0 + yy - 1, gx = tx0 + xx - 1;
        float v = 0.0f;
        if (gy >= 0 && gy < HH && gx >= 0 && gx < WW)
            v = frame[(size_t)c * HW + gy * WW + gx];
        s[c][yy][xx] = v;
    }
    __syncthreads();
    const int lx = threadIdx.x, ly = threadIdx.y;
    const int gp = (ty0 + ly) * WW + tx0 + lx;
#pragma unroll
    for (int c = 0; c < 3; ++c) {
        float m = 0.0f, m2 = 0.0f;
#pragma unroll
        for (int dy = 0; dy < 3; ++dy)
#pragma unroll
            for (int dx = 0; dx < 3; ++dx) {
                float v = s[c][ly + dy][lx + dx];
                m += v;
                m2 += v * v;
            }
        m *= (1.0f / 9.0f);
        m2 *= (1.0f / 9.0f);
        float var = fmaxf(m2 - m * m, 0.0f);
        float x = s[c][ly + 1][lx + 1];
        rec[(size_t)c * HW + gp] = (x - m) * rsqrtf(var + 1e-5f);
        rec[(size_t)(3 + c) * HW + gp] = x;
    }
}

// ---------------------------------------------------------------------------
// Bilateral: R=3 (7x7), guide = feat(8ch), dyn = src ch0..3, edge-clamped.
// ---------------------------------------------------------------------------
__global__ __launch_bounds__(256) void bilateral_kernel(
    const float* __restrict__ src, const float* __restrict__ feat,
    const float* __restrict__ scale, float* __restrict__ dst) {
    __shared__ float s_f[8][22][22];
    __shared__ float s_d[4][22][22];
    const int tx0 = blockIdx.x * 16, ty0 = blockIdx.y * 16;
    const int tid = threadIdx.y * 16 + threadIdx.x;
    for (int idx = tid; idx < 8 * 22 * 22; idx += 256) {
        int c = idx / 484;
        int r = idx % 484;
        int yy = r / 22, xx = r % 22;
        int gy = iclamp(ty0 + yy - 3, 0, HH - 1);
        int gx = iclamp(tx0 + xx - 3, 0, WW - 1);
        s_f[c][yy][xx] = feat[(size_t)c * HW + gy * WW + gx];
    }
    for (int idx = tid; idx < 4 * 22 * 22; idx += 256) {
        int c = idx / 484;
        int r = idx % 484;
        int yy = r / 22, xx = r % 22;
        int gy = iclamp(ty0 + yy - 3, 0, HH - 1);
        int gx = iclamp(tx0 + xx - 3, 0, WW - 1);
        s_d[c][yy][xx] = src[(size_t)c * HW + gy * WW + gx];
    }
    __syncthreads();
    const int lx = threadIdx.x, ly = threadIdx.y;
    float sc[8], fp[8];
#pragma unroll
    for (int c = 0; c < 8; ++c) {
        sc[c] = scale[c];
        fp[c] = s_f[c][ly + 3][lx + 3];
    }
    float a0 = 0, a1 = 0, a2 = 0, a3 = 0, wsum = 0;
    for (int dy = 0; dy < 7; ++dy) {
        for (int dx = 0; dx < 7; ++dx) {
            float d2 = 0.0f;
#pragma unroll
            for (int c = 0; c < 8; ++c) {
                float df = fp[c] - s_f[c][ly + dy][lx + dx];
                d2 += sc[c] * df * df;
            }
            float w = __expf(-d2);
            a0 += w * s_d[0][ly + dy][lx + dx];
            a1 += w * s_d[1][ly + dy][lx + dx];
            a2 += w * s_d[2][ly + dy][lx + dx];
            a3 += w * s_d[3][ly + dy][lx + dx];
            wsum += w;
        }
    }
    float inv = 1.0f / (wsum + 1e-8f);
    const int gp = (ty0 + ly) * WW + tx0 + lx;
    dst[(size_t)0 * HW + gp] = a0 * inv;
    dst[(size_t)1 * HW + gp] = a1 * inv;
    dst[(size_t)2 * HW + gp] = a2 * inv;
    dst[(size_t)3 * HW + gp] = a3 * inv;
}

// copy fixed (frame ch3..11) into hs_in ch3..11 and both filtered bufs ch4..12
__global__ __launch_bounds__(256) void copy_aux_kernel(
    const float* __restrict__ frame, float* __restrict__ hs_in,
    float* __restrict__ filtA, float* __restrict__ filtB) {
    size_t p = (size_t)blockIdx.x * 256 + threadIdx.x;
    if (p >= HW) return;
#pragma unroll
    for (int c = 0; c < 9; ++c) {
        float f = frame[(size_t)(3 + c) * HW + p];
        hs_in[(size_t)(3 + c) * HW + p] = f;
        filtA[(size_t)(4 + c) * HW + p] = f;
        filtB[(size_t)(4 + c) * HW + p] = f;
    }
}

// temporal blend into filtA ch0..3 (hidden already in filtA ch3)
__global__ __launch_bounds__(256) void blend_kernel(
    const float* __restrict__ frame, const float* __restrict__ alpha,
    const float* __restrict__ temporal, float* __restrict__ filtA, int first) {
    size_t p = (size_t)blockIdx.x * 256 + threadIdx.x;
    if (p >= HW) return;
    float a = alpha[0];
#pragma unroll
    for (int c = 0; c < 3; ++c) {
        float col = frame[(size_t)c * HW + p];
        filtA[(size_t)c * HW + p] =
            first ? col : a * col + (1.0f - a) * temporal[(size_t)c * HW + p];
    }
    if (!first) {
        float h = filtA[(size_t)3 * HW + p];
        filtA[(size_t)3 * HW + p] = a * h + (1.0f - a) * temporal[(size_t)3 * HW + p];
    }
}

// out = albedo * filtered[:3]
__global__ __launch_bounds__(256) void out_kernel(const float* __restrict__ frame,
                                                  const float* __restrict__ fin,
                                                  float* __restrict__ out) {
    size_t p = (size_t)blockIdx.x * 256 + threadIdx.x;
    if (p >= HW) return;
#pragma unroll
    for (int c = 0; c < 3; ++c)
        out[(size_t)c * HW + p] =
            frame[(size_t)(3 + c) * HW + p] * fin[(size_t)c * HW + p];
}

// ---------------------------------------------------------------------------
extern "C" void kernel_launch(void* const* d_in, const int* in_sizes, int n_in,
                              void* d_out, int out_size, void* d_ws, size_t ws_size,
                              hipStream_t stream) {
    const float* x          = (const float*)d_in[0];
    const float* alpha      = (const float*)d_in[1];
    const float* cr_w0      = (const float*)d_in[2];
    const float* cr_b0      = (const float*)d_in[3];
    const float* cr_w1      = (const float*)d_in[4];
    const float* cr_b1      = (const float*)d_in[5];
    const float* cr_w2      = (const float*)d_in[6];
    const float* cr_b2      = (const float*)d_in[7];
    const float* hs_w0      = (const float*)d_in[8];
    const float* hs_b0      = (const float*)d_in[9];
    const float* hs_w1      = (const float*)d_in[10];
    const float* hs_b1      = (const float*)d_in[11];
    const float* hs_w2      = (const float*)d_in[12];
    const float* hs_b2      = (const float*)d_in[13];
    const float* filt_w     = (const float*)d_in[14];
    const float* filt_b     = (const float*)d_in[15];
    const float* filt_scale = (const float*)d_in[16];
    float* out = (float*)d_out;
    float* ws  = (float*)d_ws;

    // workspace layout (channels of HW fp32), total 64 ch ~= 236 MB
    float* rec   = ws;                    // 6 ch
    float* bufX  = ws + (size_t)6 * HW;   // 12 ch
    float* bufY  = ws + (size_t)18 * HW;  // 12 ch
    float* filtA = ws + (size_t)30 * HW;  // 13 ch
    float* filtB = ws + (size_t)43 * HW;  // 13 ch
    float* feat  = ws + (size_t)56 * HW;  // 8 ch

    const dim3 gridT(WW / 16, HH / 16);
    const dim3 blkT(16, 16);
    const int grid1d = (HW + 255) / 256;

    for (int i = 0; i < 4; ++i) {
        const float* frame = x + (size_t)i * 12 * HW;

        // rec_in = concat(lcn(color), color)
        lcn_kernel<<<gridT, blkT, 0, stream>>>(frame, rec);

        // color-recombine chain
        conv3x3_wmma<6, 6, true><<<gridT, 128, 0, stream>>>(rec, cr_w0, cr_b0, bufX);
        conv3x3_wmma<6, 12, true><<<gridT, 128, 0, stream>>>(bufX, cr_w1, cr_b1, bufY);
        conv3x3_wmma<12, 3, false><<<gridT, 128, 0, stream>>>(bufY, cr_w2, cr_b2, bufX);

        // hs_in = concat(recombined, fixed); seed fixed into both filtered bufs
        copy_aux_kernel<<<grid1d, 256, 0, stream>>>(frame, bufX, filtA, filtB);

        // hidden-state chain (hidden written straight into filtA ch3)
        conv3x3_wmma<12, 12, true><<<gridT, 128, 0, stream>>>(bufX, hs_w0, hs_b0, bufY);
        conv3x3_wmma<12, 12, true><<<gridT, 128, 0, stream>>>(bufY, hs_w1, hs_b1, bufX);
        conv3x3_wmma<12, 1, false><<<gridT, 128, 0, stream>>>(bufX, hs_w2, hs_b2,
                                                              filtA + (size_t)3 * HW);

        // temporal blend (temporal = prev frame's final = filtB ch0..3)
        blend_kernel<<<grid1d, 256, 0, stream>>>(frame, alpha, filtB, filtA, i == 0 ? 1 : 0);

        // 5 guided-bilateral iterations, ping-pong A <-> B
        float* cur = filtA;
        float* nxt = filtB;
        for (int k = 0; k < 5; ++k) {
            conv3x3_wmma<13, 8, false><<<gridT, 128, 0, stream>>>(
                cur, filt_w + (size_t)k * 8 * 13 * 9, filt_b + (size_t)k * 8, feat);
            bilateral_kernel<<<gridT, blkT, 0, stream>>>(cur, feat,
                                                         filt_scale + (size_t)k * 8, nxt);
            float* t = cur; cur = nxt; nxt = t;
        }
        // cur == filtB after 5 swaps; also next frame's temporal source

        out_kernel<<<grid1d, 256, 0, stream>>>(frame, cur, out + (size_t)i * 3 * HW);
    }
}
